// BidirectionalAgg_77756087926983
// MI455X (gfx1250) — matrix operand order repaired and checked by reading, not compiled, not run on hardware
//
#include <hip/hip_runtime.h>
#include <hip/hip_bf16.h>
#include <math.h>

typedef __attribute__((ext_vector_type(16))) _Float16 v16h;
typedef __attribute__((ext_vector_type(8)))  _Float16 h8;
typedef __attribute__((ext_vector_type(4)))  _Float16 h4;
typedef __attribute__((ext_vector_type(2)))  __fp16   pk2;   // cvt_pkrtz return type
typedef __attribute__((ext_vector_type(8)))  float    v8f;
typedef __attribute__((ext_vector_type(2)))  float    vf2;

#define D  128   // feature dim (fixed by reference)
#define TM 64    // output rows per block
#define TK 32    // K (j) per step
#define APITCH 40  // att tile row pitch in halfs (80B: 16B-aligned, conflict-free reads)

__device__ __forceinline__ float waveReduceSum(float v) {
#pragma unroll
  for (int off = 16; off > 0; off >>= 1)
    v += __shfl_xor(v, off, 32);
  return v;
}

// fast sigmoid: v_exp_f32 + v_add + v_rcp_f32 (f16 operand downstream, rcp is fine)
__device__ __forceinline__ float sigmoidf(float z) {
  return __builtin_amdgcn_rcpf(1.0f + __expf(-z));
}

// ---------------------------------------------------------------------------
// Kernel 1: xt = logmap0(x) stored TRANSPOSED as f16 [D][n]; score vectors
// slp/srp/slc/src (bias folded into the left terms). One wave per row.
// ---------------------------------------------------------------------------
__global__ __launch_bounds__(256) void prep_kernel(
    const float* __restrict__ x,
    const float* __restrict__ w_par, const float* __restrict__ b_par,
    const float* __restrict__ w_chi, const float* __restrict__ b_chi,
    _Float16* __restrict__ xt16T,
    float* __restrict__ slp, float* __restrict__ srp,
    float* __restrict__ slc, float* __restrict__ src_, int n)
{
  __shared__ __align__(16) _Float16 tile[8][D];  // 8 rows staged for transpose

  const int wave = threadIdx.x >> 5;
  const int lane = threadIdx.x & 31;
  const int row  = blockIdx.x * 8 + wave;        // grid = n/8 exactly

  const float4 xv = reinterpret_cast<const float4*>(x + (size_t)row * D)[lane];
  float ss = waveReduceSum(xv.x*xv.x + xv.y*xv.y + xv.z*xv.z + xv.w*xv.w);
  const float norm  = sqrtf(ss);
  const float normc = fmaxf(norm, 1e-15f);
  const float xc    = fminf(normc, 1.0f - 1e-7f);
  const float at    = 0.5f * (log1pf(xc) - log1pf(-xc));   // artanh
  const float scale = at / normc;

  float4 t;
  t.x = xv.x * scale; t.y = xv.y * scale; t.z = xv.z * scale; t.w = xv.w * scale;
  h4 th; th[0] = (_Float16)t.x; th[1] = (_Float16)t.y;
         th[2] = (_Float16)t.z; th[3] = (_Float16)t.w;
  *reinterpret_cast<h4*>(&tile[wave][lane * 4]) = th;

  const float4 wpl = reinterpret_cast<const float4*>(w_par)[lane];
  const float4 wph = reinterpret_cast<const float4*>(w_par + D)[lane];
  const float4 wcl = reinterpret_cast<const float4*>(w_chi)[lane];
  const float4 wch = reinterpret_cast<const float4*>(w_chi + D)[lane];
  const float d0 = waveReduceSum(t.x*wpl.x + t.y*wpl.y + t.z*wpl.z + t.w*wpl.w);
  const float d1 = waveReduceSum(t.x*wph.x + t.y*wph.y + t.z*wph.z + t.w*wph.w);
  const float d2 = waveReduceSum(t.x*wcl.x + t.y*wcl.y + t.z*wcl.z + t.w*wcl.w);
  const float d3 = waveReduceSum(t.x*wch.x + t.y*wch.y + t.z*wch.z + t.w*wch.w);
  if (lane == 0) {
    slp[row]  = d0 + b_par[0];
    srp[row]  = d1;
    slc[row]  = d2 + b_chi[0];
    src_[row] = d3;
  }

  __syncthreads();
  // transpose 8x128 -> xt16T[d][row0..row0+7]: one coalesced-per-thread b128 store
  if (threadIdx.x < D) {
    const int d = threadIdx.x;
    h8 colv;
#pragma unroll
    for (int r = 0; r < 8; ++r) colv[r] = tile[r][d];
    *reinterpret_cast<h8*>(xt16T + (size_t)d * n + blockIdx.x * 8) = colv;
  }
}

// ---------------------------------------------------------------------------
// Kernel 2: support = (A_par + A_chi) @ xt via v_wmma_f32_16x16x32_f16.
// 256 threads (8 waves); block = 64 rows x 128 cols of output.
// Wave w: rows (w&3)*16..+15, n-tiles (w>>2)*4..+3 (C = 4 x v8f).
// A frag = two contiguous 16B LDS runs per lane; B frag = one contiguous
// 32B global load per lane from transposed xt (L2-resident).
// ---------------------------------------------------------------------------
__global__ __launch_bounds__(256) void agg_kernel(
    const float* __restrict__ adj,
    const float* __restrict__ slp, const float* __restrict__ srp,
    const float* __restrict__ slc, const float* __restrict__ src_,
    const _Float16* __restrict__ xt16T,
    float* __restrict__ support, int n)
{
  __shared__ __align__(16) _Float16 attP[TM][APITCH];  // adj[i,j]*sig_par (f16)
  __shared__ __align__(16) _Float16 attC[TM][APITCH];  // adj[j,i]*sig_chi (f16)

  const int tid  = threadIdx.x;
  const int lane = tid & 31;
  const int wave = tid >> 5;
  const int i0   = blockIdx.x * TM;

  // part-1 mapping (row-major adj, b64 pair loads): lanes sweep j-pairs
  const int p1_jh = tid & 15;                 // j-pair index 0..15
  const int p1_ib = tid >> 4;                 // 0..15
  float slp_r[4];
#pragma unroll
  for (int it = 0; it < 4; ++it) slp_r[it] = slp[i0 + p1_ib + 16 * it];

  // part-2 mapping (transposed adj): lanes sweep i -> coalesced b32 loads
  const int p2_i   = tid & 63;
  const int p2_jhb = tid >> 6;                // 0..3
  const float slc_r = slc[i0 + p2_i];

  // wave tile assignment
  const int wm   = (wave & 3) * 16;
  const int wn0  = (wave >> 2) * 4;
  const int half = lane >> 4;
  const int fm   = wm + (lane & 15);

  v8f c[4] = {};

  for (int j0 = 0; j0 < n; j0 += TK) {
    __syncthreads();   // protect LDS tiles from previous iteration's readers

    // ---- part 1: A_par tile ----
    {
      const vf2 srp2 = *reinterpret_cast<const vf2*>(srp + j0 + 2 * p1_jh);
#pragma unroll
      for (int it = 0; it < 4; ++it) {
        const int il = p1_ib + 16 * it;
        const float* ap = adj + (size_t)(i0 + il) * n + j0 + 2 * p1_jh;
        __builtin_prefetch(ap + TK, 0, 0);
        const vf2 a = __builtin_nontemporal_load(reinterpret_cast<const vf2*>(ap));
        float v0 = 0.0f, v1 = 0.0f;
        if (a[0] != 0.0f) v0 = a[0] * sigmoidf(slp_r[it] + srp2[0]);
        if (a[1] != 0.0f) v1 = a[1] * sigmoidf(slp_r[it] + srp2[1]);
        const pk2 p = __builtin_amdgcn_cvt_pkrtz(v0, v1);
        *reinterpret_cast<pk2*>(&attP[il][2 * p1_jh]) = p;
      }
    }
    // ---- part 2: A_chi tile (adj read transposed, coalesced over i) ----
#pragma unroll
    for (int s = 0; s < 4; ++s) {
      const int jh = p2_jhb + 4 * s;          // 0..15
      const int jj = 2 * jh;
      const float* ap0 = adj + (size_t)(j0 + jj) * n + i0 + p2_i;
      __builtin_prefetch(ap0 + (size_t)TK * n, 0, 0);
      const float a0 = __builtin_nontemporal_load(ap0);
      const float a1 = __builtin_nontemporal_load(ap0 + n);
      const vf2 src2 = *reinterpret_cast<const vf2*>(src_ + j0 + jj);
      float v0 = 0.0f, v1 = 0.0f;
      if (a0 != 0.0f) v0 = a0 * sigmoidf(slc_r + src2[0]);
      if (a1 != 0.0f) v1 = a1 * sigmoidf(slc_r + src2[1]);
      const pk2 p = __builtin_amdgcn_cvt_pkrtz(v0, v1);
      *reinterpret_cast<pk2*>(&attC[p2_i][jj]) = p;
    }
    __syncthreads();

    // ---- A fragment: two 16B LDS runs per buffer, summed with v_pk_add_f16 ----
    union { v16h v; h8 h[2]; } A;
    {
      const h8 pl = *reinterpret_cast<const h8*>(&attP[fm][16 * half]);
      const h8 ph = *reinterpret_cast<const h8*>(&attP[fm][16 * half + 8]);
      const h8 cl = *reinterpret_cast<const h8*>(&attC[fm][16 * half]);
      const h8 ch = *reinterpret_cast<const h8*>(&attC[fm][16 * half + 8]);
      A.h[0] = pl + cl;
      A.h[1] = ph + ch;
    }
    // ---- B fragments: contiguous 32B loads from transposed xt + WMMA ----
#pragma unroll
    for (int nt = 0; nt < 4; ++nt) {
      const int col = (wn0 + nt) * 16 + (lane & 15);
      const v16h b = *reinterpret_cast<const v16h*>(
          xt16T + (size_t)col * n + j0 + 16 * half);
      c[nt] = __builtin_amdgcn_wmma_f32_16x16x32_f16(
          false, A.v, false, b, (short)0, c[nt], false, false);
    }
  }

  // ---- epilogue: C layout VGPR r -> M = r + 8*half, N = lane&15 ----
#pragma unroll
  for (int nt = 0; nt < 4; ++nt) {
#pragma unroll
    for (int r = 0; r < 8; ++r) {
      const int row = i0 + wm + r + 8 * half;
      const int col = (wn0 + nt) * 16 + (lane & 15);
      support[(size_t)row * D + col] = c[nt][r];
    }
  }
}

// ---------------------------------------------------------------------------
// Kernel 3: out = proj(expmap0(support)) in place. ||expmap0(u)|| == tanh(||u||)
// so the proj clamp folds into a single scale. One wave per row.
// ---------------------------------------------------------------------------
__global__ __launch_bounds__(256) void final_kernel(float* __restrict__ buf, int n)
{
  const int wave = threadIdx.x >> 5;
  const int lane = threadIdx.x & 31;
  const int row  = blockIdx.x * 8 + wave;
  float4 u = reinterpret_cast<const float4*>(buf + (size_t)row * D)[lane];
  float ss = waveReduceSum(u.x*u.x + u.y*u.y + u.z*u.z + u.w*u.w);
  const float normc = fmaxf(sqrtf(ss), 1e-15f);
  const float tn    = tanhf(normc);
  const float f     = fminf(tn, 1.0f - 1e-5f) / normc;
  float4 o; o.x = u.x*f; o.y = u.y*f; o.z = u.z*f; o.w = u.w*f;
  reinterpret_cast<float4*>(buf + (size_t)row * D)[lane] = o;
}

// ---------------------------------------------------------------------------
extern "C" void kernel_launch(void* const* d_in, const int* in_sizes, int n_in,
                              void* d_out, int out_size, void* d_ws, size_t ws_size,
                              hipStream_t stream) {
  const float* x     = (const float*)d_in[0];
  const float* adj   = (const float*)d_in[1];
  const float* w_par = (const float*)d_in[2];
  const float* b_par = (const float*)d_in[3];
  const float* w_chi = (const float*)d_in[4];
  const float* b_chi = (const float*)d_in[5];
  const int n = in_sizes[0] / D;               // 8192

  char* ws = (char*)d_ws;
  _Float16* xt16T = (_Float16*)ws;             // [D][n] halfs (2 MB)
  float* slp  = (float*)(ws + (size_t)n * D * 2);
  float* srp  = slp + n;
  float* slc  = srp + n;
  float* src_ = slc + n;
  float* support = (float*)d_out;              // reuse output as scratch

  prep_kernel<<<n / 8, 256, 0, stream>>>(x, w_par, b_par, w_chi, b_chi,
                                         xt16T, slp, srp, slc, src_, n);
  agg_kernel<<<n / TM, 256, 0, stream>>>(adj, slp, srp, slc, src_,
                                         xt16T, support, n);
  final_kernel<<<n / 8, 256, 0, stream>>>(support, n);
}
